// VQVAE_24885040513214
// MI455X (gfx1250) — matrix operand order repaired and checked by reading, not compiled, not run on hardware
//
#include <hip/hip_runtime.h>
#include <hip/hip_bf16.h>

typedef __attribute__((ext_vector_type(16))) _Float16 v16h;
typedef __attribute__((ext_vector_type(8)))  float    v8f;
typedef __attribute__((ext_vector_type(8)))  int      v8i;
typedef __attribute__((ext_vector_type(2)))  _Float16 v2h;

#define BATCH 1024
#define LSEQ  2048
#define DIM   64
#define KCODE 8192
#define SLOPE 0.001f
#define BETA  0.25f

// SLOPE < 1  =>  leaky(x) == max(x, SLOPE*x)  (v_mul + v_max, no cmp/cndmask)
__device__ __forceinline__ float leaky(float v) { return fmaxf(v, SLOPE * v); }

// ---------------------------------------------------------------------------
// Kernel 1: fused encoder (conv1x1 3->64->128->64 over L) + max pool over L.
// One workgroup per batch element, 128 threads = 4 waves, 16-position tiles.
// Activations never touch LDS: layer-1 output is computed straight into the
// WMMA B-fragment registers, and the layer-2 C-fragment is converted to the
// layer-3 B-fragment with one cross-half shfl_xor(16) exchange. The compiler
// hoists all 32 weight A-fragments into the >256 VGPR file (register-resident
// weights, 1 wave/SIMD); x loads are software-pipelined one tile ahead so
// VMEM latency overlaps the 32 WMMAs per tile.
// ---------------------------------------------------------------------------
__launch_bounds__(128, 1)
__global__ void encoder_kernel(const float* __restrict__ x,
                               const float* __restrict__ W1, const float* __restrict__ b1,
                               const float* __restrict__ W2, const float* __restrict__ b2,
                               const float* __restrict__ W3, const float* __restrict__ b3,
                               float* __restrict__ zout)
{
    // W2 fragments: [mt(8)][kt(2)][lane(32)][e(16)]  (A-matrix 16x32 f16 layout)
    __shared__ __align__(32) _Float16 sW2f[8 * 2 * 32 * 16];
    // W3 fragments: [mt(4)][kt(4)][lane(32)][e(16)]
    __shared__ __align__(32) _Float16 sW3f[4 * 4 * 32 * 16];
    __shared__ float sW1[64 * 3];
    __shared__ float sb1[64], sb2[128], sb3[64];
    __shared__ float szred[4][64];

    const int b   = blockIdx.x;
    const int tid = threadIdx.x;
    const int wv  = tid >> 5;
    const int ln  = tid & 31;
    const int nn  = ln & 15;   // column / position within tile
    const int hf  = ln >> 4;   // lane half

    // ---- cooperative weight load: convert to f16, pre-swizzle A-fragments ----
    for (int i = tid; i < 8 * 2 * 32 * 16; i += 128) {
        const int e    = i & 15;
        const int lane = (i >> 4) & 31;
        const int kt   = (i >> 9) & 1;
        const int mt   = i >> 10;
        const int m    = lane & 15, half = lane >> 4;
        const int k    = kt * 32 + half * 16 + e;
        sW2f[i] = (_Float16)W2[(mt * 16 + m) * 64 + k];
    }
    for (int i = tid; i < 4 * 4 * 32 * 16; i += 128) {
        const int e    = i & 15;
        const int lane = (i >> 4) & 31;
        const int kt   = (i >> 9) & 3;
        const int mt   = i >> 11;
        const int m    = lane & 15, half = lane >> 4;
        const int k    = kt * 32 + half * 16 + e;
        sW3f[i] = (_Float16)W3[(mt * 16 + m) * 128 + k];
    }
    for (int i = tid; i < 64 * 3; i += 128) sW1[i] = W1[i];
    if (tid < 64)  { sb1[tid] = b1[tid]; sb3[tid] = b3[tid]; }
    if (tid < 128) sb2[tid] = b2[tid];
    __syncthreads();

    const float* xb = x + (size_t)b * 3 * LSEQ;

    v8f zmax[4];
#pragma unroll
    for (int mt = 0; mt < 4; ++mt)
#pragma unroll
        for (int r = 0; r < 8; ++r) zmax[mt][r] = -3.4e38f;

    // software-pipelined x loads (one tile ahead)
    int p = (0 * 4 + wv) * 16 + nn;
    float cx0 = xb[0 * LSEQ + p];
    float cx1 = xb[1 * LSEQ + p];
    float cx2 = xb[2 * LSEQ + p];

    // 2048 positions / (4 waves * 16) = 32 tiles per wave
    for (int t = 0; t < 32; ++t) {
        // issue next tile's loads before this tile's compute (branchless clamp)
        const int tn = (t + 1 < 32) ? (t + 1) : 31;
        const int pn = (tn * 4 + wv) * 16 + nn;
        const float nx0 = xb[0 * LSEQ + pn];
        const float nx1 = xb[1 * LSEQ + pn];
        const float nx2 = xb[2 * LSEQ + pn];

        // ---- layer 1 (K=3): straight into the two layer-2 B-fragments.
        // B order chosen here (same k order as A): lane (n,hf) element e  <->  K = hf*16 + e.
        v16h bf2[2];
#pragma unroll
        for (int kt = 0; kt < 2; ++kt) {
#pragma unroll
            for (int e = 0; e < 16; ++e) {
                const int c = kt * 32 + hf * 16 + e;
                const float v = leaky(fmaf(sW1[c * 3 + 0], cx0,
                                      fmaf(sW1[c * 3 + 1], cx1,
                                      fmaf(sW1[c * 3 + 2], cx2, sb1[c]))));
                bf2[kt][e] = (_Float16)v;
            }
        }

        // ---- layer 2: h2[128,16] = W2[128x64] * h1[64x16] ----
        v8f acc2[8];
#pragma unroll
        for (int mt = 0; mt < 8; ++mt) acc2[mt] = (v8f){0.f,0.f,0.f,0.f,0.f,0.f,0.f,0.f};
#pragma unroll
        for (int kt = 0; kt < 2; ++kt) {
#pragma unroll
            for (int mt = 0; mt < 8; ++mt) {
                const v16h af = *(const v16h*)&sW2f[(((mt * 2 + kt) * 32) + ln) * 16];
                acc2[mt] = __builtin_amdgcn_wmma_f32_16x16x32_f16(
                    false, af, false, bf2[kt], (short)0, acc2[mt], false, false);
            }
        }

        // ---- epilogue: bias + leaky, pack to f16 pairs, cross-half exchange.
        // C layout: lane (n,hf) VGPR r holds channel mt*16 + hf*8 + r.
        int lv[8][4], pv[8][4];
#pragma unroll
        for (int mt = 0; mt < 8; ++mt) {
#pragma unroll
            for (int rp = 0; rp < 4; ++rp) {
                const int m = mt * 16 + hf * 8 + 2 * rp;
                v2h pk;
                pk.x = (_Float16)leaky(acc2[mt][2 * rp]     + sb2[m]);
                pk.y = (_Float16)leaky(acc2[mt][2 * rp + 1] + sb2[m + 1]);
                lv[mt][rp] = __builtin_bit_cast(int, pk);
                pv[mt][rp] = __shfl_xor(lv[mt][rp], 16, 32);
            }
        }

        // ---- layer 3: h3[64,16] = W3[64x128] * h2[128x16] ----
        v8f acc3[4];
#pragma unroll
        for (int mt = 0; mt < 4; ++mt) acc3[mt] = (v8f){0.f,0.f,0.f,0.f,0.f,0.f,0.f,0.f};
#pragma unroll
        for (int kt = 0; kt < 4; ++kt) {
            // Rebuild B-fragment for this K-tile: lane (n,hf) needs channels
            // kt*32 + hf*16 + (0..15); half of them come from the partner lane.
            v8i bw;
#pragma unroll
            for (int d = 0; d < 4; ++d) {
                bw[d]     = hf ? pv[2 * kt + 1][d] : lv[2 * kt][d];
                bw[d + 4] = hf ? lv[2 * kt + 1][d] : pv[2 * kt][d];
            }
            const v16h bf = __builtin_bit_cast(v16h, bw);
#pragma unroll
            for (int mt = 0; mt < 4; ++mt) {
                const v16h af = *(const v16h*)&sW3f[(((mt * 4 + kt) * 32) + ln) * 16];
                acc3[mt] = __builtin_amdgcn_wmma_f32_16x16x32_f16(
                    false, af, false, bf, (short)0, acc3[mt], false, false);
            }
        }

        // bias + running max over positions (no activation on layer 3)
#pragma unroll
        for (int mt = 0; mt < 4; ++mt)
#pragma unroll
            for (int r = 0; r < 8; ++r)
                zmax[mt][r] = fmaxf(zmax[mt][r], acc3[mt][r] + sb3[mt * 16 + hf * 8 + r]);

        cx0 = nx0; cx1 = nx1; cx2 = nx2;
    }

    // ---- reduce max over the 16 positions (lanes within each half) ----
#pragma unroll
    for (int mt = 0; mt < 4; ++mt)
#pragma unroll
        for (int r = 0; r < 8; ++r) {
            float v = zmax[mt][r];
            v = fmaxf(v, __shfl_xor(v, 1, 32));
            v = fmaxf(v, __shfl_xor(v, 2, 32));
            v = fmaxf(v, __shfl_xor(v, 4, 32));
            v = fmaxf(v, __shfl_xor(v, 8, 32));
            zmax[mt][r] = v;
        }
    if (nn == 0) {
#pragma unroll
        for (int mt = 0; mt < 4; ++mt)
#pragma unroll
            for (int r = 0; r < 8; ++r)
                szred[wv][mt * 16 + hf * 8 + r] = zmax[mt][r];
    }
    __syncthreads();
    if (tid < 64) {
        const float v = fmaxf(fmaxf(szred[0][tid], szred[1][tid]),
                              fmaxf(szred[2][tid], szred[3][tid]));
        zout[(size_t)b * DIM + tid] = v;
    }
}

// ---------------------------------------------------------------------------
// Kernel 2: VQ argmin over 8192 codes + loss + tiny dense decoder. One block
// per batch element, 256 threads; codebook (2 MB) is L2-resident and every
// fetched line is fully consumed by the unrolled q-loop.
// ---------------------------------------------------------------------------
__launch_bounds__(256, 1)
__global__ void vq_decoder_kernel(const float* __restrict__ cb,
                                  const float* __restrict__ U1, const float* __restrict__ c1,
                                  const float* __restrict__ U2, const float* __restrict__ c2,
                                  const float* __restrict__ U3, const float* __restrict__ c3,
                                  const float* __restrict__ z,
                                  float* __restrict__ xrec,
                                  float* __restrict__ loss)
{
    __shared__ float zs[64];
    __shared__ float zq[64];
    __shared__ float g1[128];
    __shared__ float g2[64];
    __shared__ float rmin[256];
    __shared__ int   ridx[256];
    __shared__ float rsum[64];

    const int b = blockIdx.x, tid = threadIdx.x;
    if (tid < 64) zs[tid] = z[(size_t)b * DIM + tid];
    __syncthreads();

    float best = 3.4e38f; int bidx = 0;
    for (int k = tid; k < KCODE; k += 256) {
        const float4* cp = (const float4*)(cb + (size_t)k * DIM);
        float dot = 0.f, cn = 0.f;
#pragma unroll
        for (int q = 0; q < 16; ++q) {
            const float4 c4 = cp[q];
            const float* zz = &zs[q * 4];
            dot = fmaf(c4.x, zz[0], fmaf(c4.y, zz[1], fmaf(c4.z, zz[2], fmaf(c4.w, zz[3], dot))));
            cn  = fmaf(c4.x, c4.x,  fmaf(c4.y, c4.y,  fmaf(c4.z, c4.z,  fmaf(c4.w, c4.w,  cn))));
        }
        const float dist = cn - 2.f * dot;   // + ||z||^2 is constant per b
        if (dist < best) { best = dist; bidx = k; }
    }
    rmin[tid] = best; ridx[tid] = bidx;
    __syncthreads();
    for (int s = 128; s > 0; s >>= 1) {
        if (tid < s && rmin[tid + s] < rmin[tid]) {
            rmin[tid] = rmin[tid + s]; ridx[tid] = ridx[tid + s];
        }
        __syncthreads();
    }
    const int idx = ridx[0];
    if (tid < 64) {
        const float q = cb[(size_t)idx * DIM + tid];
        zq[tid] = q;
        const float d = q - zs[tid];
        rsum[tid] = d * d;
    }
    __syncthreads();
    if (tid == 0) {
        float s = 0.f;
        for (int i = 0; i < 64; ++i) s += rsum[i];
        atomicAdd(loss, s * (1.f + BETA) / (float)(BATCH * DIM));
    }
    // decoder (forward of straight-through == z_q)
    if (tid < 128) {
        float a = c1[tid];
        for (int d = 0; d < 64; ++d) a = fmaf(zq[d], U1[d * 128 + tid], a);
        g1[tid] = leaky(a);
    }
    __syncthreads();
    if (tid < 64) {
        float a = c2[tid];
        for (int i = 0; i < 128; ++i) a = fmaf(g1[i], U2[i * 64 + tid], a);
        g2[tid] = leaky(a);
    }
    __syncthreads();
    if (tid < 3) {
        float a = c3[tid];
        for (int i = 0; i < 64; ++i) a = fmaf(g2[i], U3[i * 3 + tid], a);
        xrec[(size_t)b * 3 + tid] = a;
    }
}

__global__ void zero_loss_kernel(float* loss)
{
    if (blockIdx.x == 0 && threadIdx.x == 0) *loss = 0.f;
}

extern "C" void kernel_launch(void* const* d_in, const int* in_sizes, int n_in,
                              void* d_out, int out_size, void* d_ws, size_t ws_size,
                              hipStream_t stream)
{
    const float* x  = (const float*)d_in[0];
    const float* W1 = (const float*)d_in[1];  const float* b1 = (const float*)d_in[2];
    const float* W2 = (const float*)d_in[3];  const float* b2 = (const float*)d_in[4];
    const float* W3 = (const float*)d_in[5];  const float* b3 = (const float*)d_in[6];
    const float* cb = (const float*)d_in[7];
    const float* U1 = (const float*)d_in[8];  const float* c1 = (const float*)d_in[9];
    const float* U2 = (const float*)d_in[10]; const float* c2 = (const float*)d_in[11];
    const float* U3 = (const float*)d_in[12]; const float* c3 = (const float*)d_in[13];

    float* out  = (float*)d_out;
    float* xrec = out;                       // [1024,3,1] -> 3072
    float* loss = out + BATCH * 3;           // scalar
    float* zout = out + BATCH * 3 + 1;       // [1024,64]  -> 65536

    zero_loss_kernel<<<1, 64, 0, stream>>>(loss);
    encoder_kernel<<<BATCH, 128, 0, stream>>>(x, W1, b1, W2, b2, W3, b3, zout);
    vq_decoder_kernel<<<BATCH, 256, 0, stream>>>(cb, U1, c1, U2, c2, U3, c3,
                                                 zout, xrec, loss);
}
